// MultiViewGNN_11587821764815
// MI455X (gfx1250) — matrix operand order repaired and checked by reading, not compiled, not run on hardware
//
#include <hip/hip_runtime.h>
#include <hip/hip_bf16.h>

#define FIN 256
#define NH  64

typedef __attribute__((ext_vector_type(16))) __bf16 v16bf;
typedef __attribute__((ext_vector_type(8)))  float  v8f;

__device__ __forceinline__ __bf16 f2bf(float f) {
  return (__bf16)f;   // let clang emit native bf16 cvt (v_cvt_pk_bf16_f32 when paired)
}

// ---------------- generic fill ----------------
__global__ void mv_fill(float* __restrict__ p, float v, long long n) {
  long long i = (long long)blockIdx.x * blockDim.x + threadIdx.x;
  if (i < n) p[i] = v;
}

// ---------------- degree init / accum / rsqrt ----------------
__global__ void mv_init_deg(float* __restrict__ d1, float* __restrict__ d2, int n) {
  int i = blockIdx.x * blockDim.x + threadIdx.x;
  if (i < n) { d1[i] = 1.0f; d2[i] = 1.0f; }   // self-loop contribution
}

__global__ void mv_deg_accum(const int* __restrict__ ei, const float* __restrict__ w,
                             float* __restrict__ d1, float* __restrict__ d2, int e) {
  int i = blockIdx.x * blockDim.x + threadIdx.x;
  if (i >= e) return;
  int d = ei[e + i];                            // dst row of edge_index [2,E]
  atomicAdd(&d1[d], w[i]);
  atomicAdd(&d2[d], 1.0f);
}

__global__ void mv_rsqrt2(float* __restrict__ d1, float* __restrict__ d2, int n) {
  int i = blockIdx.x * blockDim.x + threadIdx.x;
  if (i >= n) return;
  float a = d1[i], b = d2[i];
  d1[i] = a > 0.0f ? rsqrtf(a) : 0.0f;
  d2[i] = b > 0.0f ? rsqrtf(b) : 0.0f;
}

// ---------------- WMMA GEMM: xw = x @ W1  ([N,256] @ [256,64]) ----------------
// 128 threads = 4 waves. Block handles 64 rows (4 row-tiles per wave), wave w
// owns 16-col tile w. W1 staged in LDS, pre-swizzled into 32B B-fragments so the
// inner loop reads each fragment with one 32-byte ds load.
__global__ void __launch_bounds__(128)
mv_gemm_xw(const float* __restrict__ x, const float* __restrict__ W,
           float* __restrict__ xw, int n) {
  __shared__ __bf16 ldsB[FIN * NH];   // 16384 bf16 = 32 KB, fragment-swizzled

  const int wave = threadIdx.x >> 5;
  const int lane = threadIdx.x & 31;
  const int row  = lane & 15;
  const int hi   = lane >> 4;          // lane half selects K sub-range
  const int m0   = blockIdx.x << 6;    // 64 rows per block

  // ---- cooperative fill of ldsB, swizzled:
  // element (k, c):  kt=k>>5, j=(k&7)|((k>>1)&8), lane' = ((k>>3)&1)*16 + (c&15), wave'=c>>4
  // addr = (((kt*4 + wave')*32 + lane')*16 + j)
  for (int q = threadIdx.x; q < (FIN * NH) / 4; q += 128) {
    int g = q << 2;                    // global element index (row-major [K][64])
    int k = g >> 6;
    int c = g & 63;                    // multiple of 4
    float4 wv = *reinterpret_cast<const float4*>(W + g);
    int kt = k >> 5, lh = (k >> 3) & 1, j = (k & 7) | ((k >> 1) & 8);
    int base = ((kt * 4 + (c >> 4)) * 32 + lh * 16 + (c & 15)) * 16 + j;
    ldsB[base]      = f2bf(wv.x);      // c+0..c+3 share wave' (c&63 quad-aligned)
    ldsB[base + 16] = f2bf(wv.y);
    ldsB[base + 32] = f2bf(wv.z);
    ldsB[base + 48] = f2bf(wv.w);
  }
  __syncthreads();

  // ---- A row pointers (clamped for the partial last block)
  const float* xr[4];
#pragma unroll
  for (int i = 0; i < 4; ++i) {
    int m = m0 + i * 16 + row;
    if (m >= n) m = n - 1;
    xr[i] = x + (long long)m * FIN + hi * 8;
  }

  v8f acc[4] = {};
  for (int kt = 0; kt < FIN / 32; ++kt) {
    // B fragment: single 32-byte LDS read (2x ds_load_b128)
    v16bf b = *reinterpret_cast<const v16bf*>(
        &ldsB[((kt * 4 + wave) * 32 + lane) * 16]);
#pragma unroll
    for (int i = 0; i < 4; ++i) {
      const float* p = xr[i] + kt * 32;
      __builtin_prefetch(p + 64, 0, 0);          // global_prefetch_b8 two K tiles ahead
      const float4* p4 = reinterpret_cast<const float4*>(p);
      float4 a0 = p4[0], a1 = p4[1], a2 = p4[4], a3 = p4[5];
      v16bf a;
      a[0]=f2bf(a0.x);  a[1]=f2bf(a0.y);  a[2]=f2bf(a0.z);  a[3]=f2bf(a0.w);
      a[4]=f2bf(a1.x);  a[5]=f2bf(a1.y);  a[6]=f2bf(a1.z);  a[7]=f2bf(a1.w);
      a[8]=f2bf(a2.x);  a[9]=f2bf(a2.y);  a[10]=f2bf(a2.z); a[11]=f2bf(a2.w);
      a[12]=f2bf(a3.x); a[13]=f2bf(a3.y); a[14]=f2bf(a3.z); a[15]=f2bf(a3.w);
      acc[i] = __builtin_amdgcn_wmma_f32_16x16x32_bf16(false, a, false, b,
                                                       (short)0, acc[i], false, false);
    }
  }

  // ---- store; bounds check is uniform per 16-row tile (scalar branch)
  const int n0 = wave << 4;
#pragma unroll
  for (int i = 0; i < 4; ++i) {
    int rt0 = m0 + i * 16;
    float* o = xw + (long long)(rt0 + hi * 8) * NH + n0 + row;
    if (rt0 + 16 <= n) {
#pragma unroll
      for (int r = 0; r < 8; ++r) o[r * NH] = acc[i][r];
    } else {
#pragma unroll
      for (int r = 0; r < 8; ++r)
        if (rt0 + hi * 8 + r < n) o[r * NH] = acc[i][r];
    }
  }
}

// ---------------- edge scatter for conv1: 64 threads per edge ----------------
__global__ void mv_edge_scatter1(const int* __restrict__ ei, const float* __restrict__ w,
                                 const float* __restrict__ dinv, const float* __restrict__ xw,
                                 float* __restrict__ scat, int e) {
  long long t = (long long)blockIdx.x * blockDim.x + threadIdx.x;
  int edge = (int)(t >> 6);
  int f    = (int)(t & 63);
  if (edge >= e) return;
  int s = ei[edge], d = ei[e + edge];
  float norm = dinv[s] * w[edge] * dinv[d];
  atomicAdd(&scat[(long long)d * NH + f], xw[(long long)s * NH + f] * norm);
}

// ---------------- h = relu(scat + xw*dinv^2 + b1); feat += h ----------------
__global__ void mv_finish1(const float* __restrict__ scat, const float* __restrict__ xw,
                           const float* __restrict__ dinv, const float* __restrict__ b1,
                           float* __restrict__ h, float* __restrict__ feat, long long n64) {
  long long t = (long long)blockIdx.x * blockDim.x + threadIdx.x;
  if (t >= n64) return;
  int v = (int)(t >> 6), f = (int)(t & 63);
  float di = dinv[v];
  float val = scat[t] + xw[t] * di * di + b1[f];
  val = fmaxf(val, 0.0f);
  h[t] = val;
  feat[t] += val;
}

// ---------------- hw = h @ W2  (64 -> 1 matvec) ----------------
__global__ void mv_matvec(const float* __restrict__ h, const float* __restrict__ W2,
                          float* __restrict__ hw, int n) {
  int v = blockIdx.x * blockDim.x + threadIdx.x;
  if (v >= n) return;
  const float* hr = h + (long long)v * NH;
  float s = 0.0f;
#pragma unroll
  for (int f = 0; f < NH; ++f) s = fmaf(hr[f], W2[f], s);
  hw[v] = s;
}

// ---------------- scalar edge scatter for conv2 ----------------
__global__ void mv_edge_scatter2(const int* __restrict__ ei, const float* __restrict__ dinv,
                                 const float* __restrict__ hw, float* __restrict__ scat, int e) {
  int i = blockIdx.x * blockDim.x + threadIdx.x;
  if (i >= e) return;
  int s = ei[i], d = ei[e + i];
  atomicAdd(&scat[d], hw[s] * dinv[s] * dinv[d]);
}

// ---------------- out += scat + hw*dinv^2 + b2 ----------------
__global__ void mv_finish2(const float* __restrict__ scat, const float* __restrict__ hw,
                           const float* __restrict__ dinv, const float* __restrict__ b2,
                           float* __restrict__ out, int n) {
  int v = blockIdx.x * blockDim.x + threadIdx.x;
  if (v >= n) return;
  float di = dinv[v];
  out[v] += scat[v] + hw[v] * di * di + b2[0];
}

extern "C" void kernel_launch(void* const* d_in, const int* in_sizes, int n_in,
                              void* d_out, int out_size, void* d_ws, size_t ws_size,
                              hipStream_t stream) {
  (void)n_in; (void)out_size; (void)ws_size;
  const int N = in_sizes[0] / FIN;       // 50000
  const int E = in_sizes[1] / 2;         // 800000

  const float* x = (const float*)d_in[0];
  float* out  = (float*)d_out;           // [0..N): x_flat, [N..N+64N): features
  float* feat = out + N;

  // workspace layout (floats)
  float* ws    = (float*)d_ws;
  float* dinv1 = ws;                       // N
  float* dinv2 = ws + (long long)N;        // N
  float* hw    = ws + 2LL * N;             // N
  float* scat2 = ws + 3LL * N;             // N
  float* xw    = ws + 4LL * N;             // 64N
  float* scat1 = ws + 4LL * N + 64LL * N;  // 64N
  float* h     = ws + 4LL * N + 128LL * N; // 64N

  const long long n64 = (long long)N * NH;

  // zero the output (accumulated across branches)
  {
    long long tot = (long long)N + n64;
    mv_fill<<<(unsigned)((tot + 255) / 256), 256, 0, stream>>>(out, 0.0f, tot);
  }

  const int gN   = (N + 255) / 256;
  const int gE   = (E + 255) / 256;
  const int gN64 = (int)((n64 + 255) / 256);
  const int gE64 = (int)(((long long)E * NH + 255) / 256);
  const int gMT  = (N + 63) / 64;          // 64 rows per block

  for (int v = 0; v < 3; ++v) {
    const int*   ei = (const int*)  d_in[1 + v];
    const float* ew = (const float*)d_in[4 + v];
    const float* W1 = (const float*)d_in[7 + 4 * v];
    const float* b1 = (const float*)d_in[8 + 4 * v];
    const float* W2 = (const float*)d_in[9 + 4 * v];
    const float* b2 = (const float*)d_in[10 + 4 * v];

    mv_init_deg<<<gN, 256, 0, stream>>>(dinv1, dinv2, N);
    mv_deg_accum<<<gE, 256, 0, stream>>>(ei, ew, dinv1, dinv2, E);
    mv_rsqrt2<<<gN, 256, 0, stream>>>(dinv1, dinv2, N);

    mv_gemm_xw<<<gMT, 128, 0, stream>>>(x, W1, xw, N);

    mv_fill<<<gN64, 256, 0, stream>>>(scat1, 0.0f, n64);
    mv_fill<<<gN, 256, 0, stream>>>(scat2, 0.0f, (long long)N);

    mv_edge_scatter1<<<gE64, 256, 0, stream>>>(ei, ew, dinv1, xw, scat1, E);
    mv_finish1<<<gN64, 256, 0, stream>>>(scat1, xw, dinv1, b1, h, feat, n64);

    mv_matvec<<<gN, 256, 0, stream>>>(h, W2, hw, N);
    mv_edge_scatter2<<<gE, 256, 0, stream>>>(ei, dinv2, hw, scat2, E);
    mv_finish2<<<gN, 256, 0, stream>>>(scat2, hw, dinv2, b2, out, N);
  }
}
